// Attention_65541200937125
// MI455X (gfx1250) — matrix operand order repaired and checked by reading, not compiled, not run on hardware
//
#include <hip/hip_runtime.h>
#include <hip/hip_bf16.h>
#include <stdint.h>

// ---------------- types ----------------
typedef __attribute__((ext_vector_type(16))) __bf16          v16bf;
typedef __attribute__((ext_vector_type(16))) unsigned short  v16us;
typedef __attribute__((ext_vector_type(8)))  unsigned short  v8us;
typedef __attribute__((ext_vector_type(4)))  unsigned short  v4us;
typedef __attribute__((ext_vector_type(8)))  float           v8f;
typedef __attribute__((ext_vector_type(4)))  int             v4i;
typedef unsigned short us_t;

#define AS1 __attribute__((address_space(1)))
#define AS3 __attribute__((address_space(3)))

// Async global->LDS path (CDNA5). Guarded so a missing builtin degrades to
// the register-staged copy instead of breaking the build.
#if defined(__gfx1250__) && __has_builtin(__builtin_amdgcn_global_load_async_to_lds_b128)
#define HAS_ASYNC_LDS 1
#endif

// ---------------- problem constants ----------------
#define DIMC   384
#define NHEAD  6
#define HD     64
#define SEQ    2048
#define BATCH  8
#define ROWS   (BATCH*SEQ)     // 16384
#define C3     (3*DIMC)        // 1152
#define QSCALE 0.125f          // 1/sqrt(64)

// ---------------- helpers ----------------
__device__ __forceinline__ unsigned short f2bf(float f) {
  unsigned u = __float_as_uint(f);
  unsigned r = u + 0x7FFFu + ((u >> 16) & 1u);   // RNE
  return (unsigned short)(r >> 16);
}

__device__ __forceinline__ v16bf cat16(v8us lo, v8us hi) {
  union { v16us u; v16bf b; } r;
#pragma unroll
  for (int i = 0; i < 8; ++i) { r.u[i] = lo[i]; r.u[i + 8] = hi[i]; }
  return r.b;
}

__device__ __forceinline__ v8f wmma_bf16(v16bf a, v16bf b, v8f c) {
  // D = A(16x32 bf16) * B(32x16 bf16) + C(16x16 f32)
  return __builtin_amdgcn_wmma_f32_16x16x32_bf16(
      false, a, false, b, (short)0, c, false, false);
}

__device__ __forceinline__ v8f v8f_zero() {
  v8f z = {0.f, 0.f, 0.f, 0.f, 0.f, 0.f, 0.f, 0.f};
  return z;
}

// 16-byte global -> LDS copy (async on CDNA5, tracked by ASYNCcnt)
__device__ __forceinline__ void cp16_g2l(us_t* lds, const us_t* g) {
#ifdef HAS_ASYNC_LDS
  __builtin_amdgcn_global_load_async_to_lds_b128(
      (AS1 v4i*)g, (AS3 v4i*)lds, 0, 0);
#else
  *(v8us*)lds = *(const v8us*)g;
#endif
}

__device__ __forceinline__ void wait_async0() {
#ifdef HAS_ASYNC_LDS
#if __has_builtin(__builtin_amdgcn_s_wait_asynccnt)
  __builtin_amdgcn_s_wait_asynccnt(0);
#else
  __asm__ volatile("s_wait_asynccnt 0" ::: "memory");
#endif
#endif
}

// ---------------- kernel 0: f32 -> bf16 convert (x4 vectorized) ----------------
__global__ void cvt_f32_bf16(const float* __restrict__ src,
                             us_t* __restrict__ dst, int n4) {
  int i = blockIdx.x * blockDim.x + threadIdx.x;
  if (i < n4) {
    float4 f = ((const float4*)src)[i];
    v4us r;
    r[0] = f2bf(f.x); r[1] = f2bf(f.y); r[2] = f2bf(f.z); r[3] = f2bf(f.w);
    ((v4us*)dst)[i] = r;
  }
}

// ---------------- kernel 1: QKV GEMM + bias + scale + layout ----------------
// A: [ROWS, DIMC] bf16 ; W: [C3, DIMC] bf16 (row = output col, K contiguous)
// Q,K: [B,H,N,D] bf16 ; Vt: [B,H,D,N] bf16 (transposed for the PV GEMM)
#define LDA 40
#define LDB 40
__global__ __launch_bounds__(256) void qkv_gemm_kernel(
    const us_t* __restrict__ A, const us_t* __restrict__ W,
    const float* __restrict__ bias,
    us_t* __restrict__ Q, us_t* __restrict__ K, us_t* __restrict__ Vt) {
  __shared__ us_t sA[128 * LDA];
  __shared__ us_t sB[64 * LDB];
  const int tid = threadIdx.x, lane = tid & 31, wid = tid >> 5;
  const int wm = wid & 3, wn = wid >> 2;           // 4x2 wave grid
  const int m0 = blockIdx.y * 128;
  const int n0 = blockIdx.x * 64;

  v8f acc[2][2];
#pragma unroll
  for (int i = 0; i < 2; ++i)
#pragma unroll
    for (int j = 0; j < 2; ++j) acc[i][j] = v8f_zero();

  const int arow = tid >> 1, ac0 = (tid & 1) * 16;   // 2 thr/row, 128 rows
  const int brow = tid >> 2, bc0 = (tid & 3) * 8;    // 4 thr/row, 64 rows
  const us_t* Ap = A + (size_t)(m0 + arow) * DIMC;
  const us_t* Wp = W + (size_t)(n0 + brow) * DIMC;

  const int ar = lane & 15, kh8 = (lane >> 4) * 8;    // A-frag mapping
  const int bcn = lane & 15, bk16 = (lane >> 4) * 16; // B-frag mapping

  for (int kk = 0; kk < DIMC; kk += 32) {
    __syncthreads();                         // previous tile fully consumed
    cp16_g2l(&sA[arow * LDA + ac0],     Ap + kk + ac0);
    cp16_g2l(&sA[arow * LDA + ac0 + 8], Ap + kk + ac0 + 8);
    cp16_g2l(&sB[brow * LDB + bc0],     Wp + kk + bc0);
    if (kk + 32 < DIMC) {
      __builtin_prefetch(Ap + kk + 32 + ac0, 0, 1);
      __builtin_prefetch(Wp + kk + 32 + bc0, 0, 1);
    }
    wait_async0();
    __syncthreads();                         // tile visible to all waves

    v16bf afr[2], bfr[2];
#pragma unroll
    for (int t = 0; t < 2; ++t) {
      int r = wm * 32 + t * 16 + ar;
      afr[t] = cat16(*(const v8us*)&sA[r * LDA + kh8],
                     *(const v8us*)&sA[r * LDA + 16 + kh8]);
      int c = wn * 32 + t * 16 + bcn;
      bfr[t] = cat16(*(const v8us*)&sB[c * LDB + kh8],
                     *(const v8us*)&sB[c * LDB + 16 + kh8]);
    }
#pragma unroll
    for (int ti = 0; ti < 2; ++ti)
#pragma unroll
      for (int tj = 0; tj < 2; ++tj)
        acc[ti][tj] = wmma_bf16(afr[ti], bfr[tj], acc[ti][tj]);
  }

  // epilogue: bias, q-scale, scatter into Q / K / Vt
#pragma unroll
  for (int ti = 0; ti < 2; ++ti)
#pragma unroll
    for (int tj = 0; tj < 2; ++tj) {
      int gcol = n0 + wn * 32 + tj * 16 + (lane & 15);
      float bv = bias[gcol];
      int t  = gcol / DIMC;
      int hh = (gcol % DIMC) / HD;
      int dd = gcol % HD;
#pragma unroll
      for (int v = 0; v < 8; ++v) {
        int grow = m0 + wm * 32 + ti * 16 + v + 8 * (lane >> 4);
        int bb = grow >> 11, nn = grow & (SEQ - 1);
        float val = acc[ti][tj][v] + bv;
        if (t == 0) {
          Q[(((size_t)bb * NHEAD + hh) * SEQ + nn) * HD + dd] = f2bf(val * QSCALE);
        } else if (t == 1) {
          K[(((size_t)bb * NHEAD + hh) * SEQ + nn) * HD + dd] = f2bf(val);
        } else {
          Vt[(((size_t)bb * NHEAD + hh) * HD + dd) * SEQ + nn] = f2bf(val);
        }
      }
    }
}

// ---------------- kernel 2: flash attention (double-buffered async K/V) ----
// O[B,N,C] bf16 ; each wave owns 16 query rows, block = 128 rows of one (b,h)
#define SKLD 72
#define SVLD 40
#define SPLD 40
#define SKBUF (32 * SKLD)
#define SVBUF (64 * SVLD)
__global__ __launch_bounds__(256) void attn_kernel(
    const us_t* __restrict__ Q, const us_t* __restrict__ K,
    const us_t* __restrict__ Vt, us_t* __restrict__ O) {
  __shared__ us_t sK[2 * SKBUF];          // 2 x (32 keys x 64 d)
  __shared__ us_t sV[2 * SVBUF];          // 2 x (64 d x 32 keys, transposed)
  __shared__ us_t sP[8 * 16 * SPLD];      // per-wave 16x32 probs
  const int tid = threadIdx.x, lane = tid & 31, wid = tid >> 5;
  const int qblk = blockIdx.x;            // 0..15
  const int bh = blockIdx.y;              // 0..47
  const size_t qkbase = (size_t)bh * SEQ * HD;
  const size_t vbase  = (size_t)bh * HD * SEQ;

  const int kh8 = (lane >> 4) * 8;
  const int q0 = qblk * 128 + wid * 16;
  const int qr = q0 + (lane & 15);

  v16bf qf[2];
#pragma unroll
  for (int ks = 0; ks < 2; ++ks)
    qf[ks] = cat16(*(const v8us*)(Q + qkbase + (size_t)qr * HD + ks * 32 + kh8),
                   *(const v8us*)(Q + qkbase + (size_t)qr * HD + ks * 32 + 16 + kh8));

  float mrow[8], lrow[8];
#pragma unroll
  for (int v = 0; v < 8; ++v) { mrow[v] = -1e30f; lrow[v] = 0.f; }
  v8f Oacc[4];
#pragma unroll
  for (int j = 0; j < 4; ++j) Oacc[j] = v8f_zero();

  const int krow = tid >> 3, kcs = (tid & 7) * 8;  // sK staging (32x64)
  const int vrow = tid >> 2, vcs = (tid & 3) * 8;  // sV staging (64x32)
  us_t* pb = &sP[wid * 16 * SPLD];
  const int bcn = lane & 15, bk16 = (lane >> 4) * 16;

  auto issue_tile = [&](int kt, int buf) {
    cp16_g2l(&sK[buf * SKBUF + krow * SKLD + kcs],
             K + qkbase + (size_t)(kt * 32 + krow) * HD + kcs);
    cp16_g2l(&sV[buf * SVBUF + vrow * SVLD + vcs],
             Vt + vbase + (size_t)vrow * SEQ + kt * 32 + vcs);
  };

  issue_tile(0, 0);                        // prologue prefetch into buffer 0

  for (int kt = 0; kt < SEQ / 32; ++kt) {
    const int cur = kt & 1;
    wait_async0();                         // this wave's async writes landed
    __syncthreads();                       // everyone's tile-kt data visible,
                                           // and tile kt-1 fully consumed
    if (kt + 1 < SEQ / 32) issue_tile(kt + 1, 1 - cur);  // overlap w/ compute

    const us_t* sKc = &sK[cur * SKBUF];
    const us_t* sVc = &sV[cur * SVBUF];

    // S = Q * K^T  (16 rows x 32 keys, as two 16x16 tiles)
    v8f S0 = v8f_zero(), S1 = v8f_zero();
#pragma unroll
    for (int ks = 0; ks < 2; ++ks) {
      v16bf kf0 = cat16(*(const v8us*)&sKc[bcn * SKLD + ks * 32 + kh8],
                        *(const v8us*)&sKc[bcn * SKLD + ks * 32 + 16 + kh8]);
      S0 = wmma_bf16(qf[ks], kf0, S0);
      v16bf kf1 = cat16(*(const v8us*)&sKc[(16 + bcn) * SKLD + ks * 32 + kh8],
                        *(const v8us*)&sKc[(16 + bcn) * SKLD + ks * 32 + 16 + kh8]);
      S1 = wmma_bf16(qf[ks], kf1, S1);
    }

    // online softmax across the 32 keys (row stats live in 16-lane groups)
    float alpha[8];
#pragma unroll
    for (int v = 0; v < 8; ++v) {
      float mx = fmaxf(S0[v], S1[v]);
#pragma unroll
      for (int msk = 1; msk < 16; msk <<= 1) mx = fmaxf(mx, __shfl_xor(mx, msk, 32));
      float mn = fmaxf(mrow[v], mx);
      alpha[v] = __expf(mrow[v] - mn);
      mrow[v] = mn;
      float p0 = __expf(S0[v] - mn);
      float p1 = __expf(S1[v] - mn);
      S0[v] = p0; S1[v] = p1;
      float rs = p0 + p1;
#pragma unroll
      for (int msk = 1; msk < 16; msk <<= 1) rs += __shfl_xor(rs, msk, 32);
      lrow[v] = lrow[v] * alpha[v] + rs;
    }
#pragma unroll
    for (int j = 0; j < 4; ++j)
#pragma unroll
      for (int v = 0; v < 8; ++v) Oacc[j][v] *= alpha[v];

    // C-layout -> A-layout for P via per-wave LDS patch
#pragma unroll
    for (int v = 0; v < 8; ++v) {
      int pr = v + 8 * (lane >> 4);
      pb[pr * SPLD + (lane & 15)]      = f2bf(S0[v]);
      pb[pr * SPLD + 16 + (lane & 15)] = f2bf(S1[v]);
    }
    __asm__ volatile("s_wait_dscnt 0" ::: "memory");
    v16bf pf = cat16(*(const v8us*)&pb[(lane & 15) * SPLD + kh8],
                     *(const v8us*)&pb[(lane & 15) * SPLD + 16 + kh8]);

    // O += P * V
#pragma unroll
    for (int j = 0; j < 4; ++j) {
      v16bf vf = cat16(*(const v8us*)&sVc[(j * 16 + bcn) * SVLD + kh8],
                       *(const v8us*)&sVc[(j * 16 + bcn) * SVLD + 16 + kh8]);
      Oacc[j] = wmma_bf16(pf, vf, Oacc[j]);
    }
  }

  // normalize + write [B,N,C] bf16
  int b = bh / NHEAD, h = bh % NHEAD;
  float invl[8];
#pragma unroll
  for (int v = 0; v < 8; ++v) invl[v] = 1.0f / lrow[v];
#pragma unroll
  for (int j = 0; j < 4; ++j)
#pragma unroll
    for (int v = 0; v < 8; ++v) {
      int nn = q0 + v + 8 * (lane >> 4);
      float val = Oacc[j][v] * invl[v];
      O[((size_t)b * SEQ + nn) * DIMC + h * HD + j * 16 + (lane & 15)] = f2bf(val);
    }
}

// ---------------- kernel 3: output projection ----------------
// A: attn [ROWS, DIMC] bf16 ; W: [DIMC, DIMC] bf16 ; out: [ROWS, DIMC] f32
__global__ __launch_bounds__(256) void proj_gemm_kernel(
    const us_t* __restrict__ A, const us_t* __restrict__ W,
    const float* __restrict__ bias, float* __restrict__ out) {
  __shared__ us_t sA[128 * LDA];
  __shared__ us_t sB[64 * LDB];
  const int tid = threadIdx.x, lane = tid & 31, wid = tid >> 5;
  const int wm = wid & 3, wn = wid >> 2;
  const int m0 = blockIdx.y * 128;
  const int n0 = blockIdx.x * 64;

  v8f acc[2][2];
#pragma unroll
  for (int i = 0; i < 2; ++i)
#pragma unroll
    for (int j = 0; j < 2; ++j) acc[i][j] = v8f_zero();

  const int arow = tid >> 1, ac0 = (tid & 1) * 16;
  const int brow = tid >> 2, bc0 = (tid & 3) * 8;
  const us_t* Ap = A + (size_t)(m0 + arow) * DIMC;
  const us_t* Wp = W + (size_t)(n0 + brow) * DIMC;

  const int ar = lane & 15, kh8 = (lane >> 4) * 8;
  const int bcn = lane & 15, bk16 = (lane >> 4) * 16;

  for (int kk = 0; kk < DIMC; kk += 32) {
    __syncthreads();
    cp16_g2l(&sA[arow * LDA + ac0],     Ap + kk + ac0);
    cp16_g2l(&sA[arow * LDA + ac0 + 8], Ap + kk + ac0 + 8);
    cp16_g2l(&sB[brow * LDB + bc0],     Wp + kk + bc0);
    if (kk + 32 < DIMC) {
      __builtin_prefetch(Ap + kk + 32 + ac0, 0, 1);
      __builtin_prefetch(Wp + kk + 32 + bc0, 0, 1);
    }
    wait_async0();
    __syncthreads();

    v16bf afr[2], bfr[2];
#pragma unroll
    for (int t = 0; t < 2; ++t) {
      int r = wm * 32 + t * 16 + ar;
      afr[t] = cat16(*(const v8us*)&sA[r * LDA + kh8],
                     *(const v8us*)&sA[r * LDA + 16 + kh8]);
      int c = wn * 32 + t * 16 + bcn;
      bfr[t] = cat16(*(const v8us*)&sB[c * LDB + kh8],
                     *(const v8us*)&sB[c * LDB + 16 + kh8]);
    }
#pragma unroll
    for (int ti = 0; ti < 2; ++ti)
#pragma unroll
      for (int tj = 0; tj < 2; ++tj)
        acc[ti][tj] = wmma_bf16(afr[ti], bfr[tj], acc[ti][tj]);
  }

#pragma unroll
  for (int ti = 0; ti < 2; ++ti)
#pragma unroll
    for (int tj = 0; tj < 2; ++tj) {
      int gcol = n0 + wn * 32 + tj * 16 + (lane & 15);
      float bv = bias[gcol];
#pragma unroll
      for (int v = 0; v < 8; ++v) {
        int grow = m0 + wm * 32 + ti * 16 + v + 8 * (lane >> 4);
        out[(size_t)grow * DIMC + gcol] = acc[ti][tj][v] + bv;
      }
    }
}

// ---------------- launch ----------------
extern "C" void kernel_launch(void* const* d_in, const int* in_sizes, int n_in,
                              void* d_out, int out_size, void* d_ws, size_t ws_size,
                              hipStream_t stream) {
  const float* x      = (const float*)d_in[0];
  const float* qkv_w  = (const float*)d_in[1];
  const float* qkv_b  = (const float*)d_in[2];
  const float* proj_w = (const float*)d_in[3];
  const float* proj_b = (const float*)d_in[4];
  float* out = (float*)d_out;
  (void)in_sizes; (void)n_in; (void)out_size; (void)ws_size;

  char* ws = (char*)d_ws;
  size_t off = 0;
  auto wsalloc = [&](size_t bytes) -> void* {
    void* p = ws + off;
    off = (off + bytes + 255) & ~(size_t)255;
    return p;
  };
  us_t* xb    = (us_t*)wsalloc((size_t)ROWS * DIMC * 2);
  us_t* wqkv  = (us_t*)wsalloc((size_t)C3 * DIMC * 2);
  us_t* wproj = (us_t*)wsalloc((size_t)DIMC * DIMC * 2);
  us_t* Qb    = (us_t*)wsalloc((size_t)BATCH * NHEAD * SEQ * HD * 2);
  us_t* Kb    = (us_t*)wsalloc((size_t)BATCH * NHEAD * SEQ * HD * 2);
  us_t* Vtb   = (us_t*)wsalloc((size_t)BATCH * NHEAD * HD * SEQ * 2);
  us_t* attnb = (us_t*)wsalloc((size_t)ROWS * DIMC * 2);

  // 0) f32 -> bf16 conversions
  {
    int n4 = ROWS * DIMC / 4;
    cvt_f32_bf16<<<(n4 + 255) / 256, 256, 0, stream>>>(x, xb, n4);
    n4 = C3 * DIMC / 4;
    cvt_f32_bf16<<<(n4 + 255) / 256, 256, 0, stream>>>(qkv_w, wqkv, n4);
    n4 = DIMC * DIMC / 4;
    cvt_f32_bf16<<<(n4 + 255) / 256, 256, 0, stream>>>(proj_w, wproj, n4);
  }
  // 1) QKV GEMM (+bias, q-scale, [B,H,N,D]/[B,H,D,N] layout)
  qkv_gemm_kernel<<<dim3(C3 / 64, ROWS / 128), 256, 0, stream>>>(
      xb, wqkv, qkv_b, Qb, Kb, Vtb);
  // 2) flash attention
  attn_kernel<<<dim3(SEQ / 128, BATCH * NHEAD), 256, 0, stream>>>(
      Qb, Kb, Vtb, attnb);
  // 3) output projection (+bias) -> f32
  proj_gemm_kernel<<<dim3(DIMC / 64, ROWS / 128), 256, 0, stream>>>(
      attnb, wproj, proj_b, out);
}
